// SimpleGAT_39702677684792
// MI455X (gfx1250) — matrix operand order repaired and checked by reading, not compiled, not run on hardware
//
#include <hip/hip_runtime.h>
#include <cstdint>
#include <cstddef>

// ---------------------------------------------------------------------------
// Model constants (match the reference)
// ---------------------------------------------------------------------------
#define CAT      128      // HEADS*HID (also K of every GEMM in this model)
#define HEADS    4
#define HID      32
#define NGRAPH   64
#define SLOPE    0.2f
#define GN_EPS   1e-5f

typedef __attribute__((ext_vector_type(16))) _Float16 v16h;
typedef __attribute__((ext_vector_type(8)))  float    v8f;

// ---------------------------------------------------------------------------
// Helpers
// ---------------------------------------------------------------------------
__device__ __forceinline__ float lrelu(float x) { return x > 0.f ? x : SLOPE * x; }

// Order-preserving float <-> u32 map so float segment-max can use atomicMax(u32).
__device__ __forceinline__ unsigned enc_ord(float f) {
    unsigned b = __float_as_uint(f);
    return (b & 0x80000000u) ? ~b : (b | 0x80000000u);
}
__device__ __forceinline__ float dec_ord(unsigned u) {
    unsigned b = (u & 0x80000000u) ? (u & 0x7FFFFFFFu) : ~u;
    return __uint_as_float(b);
}

// ---------------------------------------------------------------------------
// WMMA GEMM:  C[M,Nc] = A[M,128] * B[Nc,128]^T (+ bias[col])
// One wave32 computes a 16x64 output strip (4 accumulators): the A fragment
// is loaded once per k-step and reused across 4 WMMAs. K=128 fully unrolled
// -> 16 static v_wmma_f32_16x16x32_f16 per kernel.
// M multiple of 16, Nc multiple of 64.
// ---------------------------------------------------------------------------
__global__ __launch_bounds__(256) void wmma_gemm_nt_kernel(
    const float* __restrict__ A, const float* __restrict__ B,
    const float* __restrict__ bias, float* __restrict__ C,
    int M, int Nc)
{
    const int lane    = threadIdx.x & 31;
    const int wave    = threadIdx.x >> 5;
    const int groupsN = Nc >> 6;                       // 64-col groups
    const int numGrp  = (M >> 4) * groupsN;
    const int grp     = blockIdx.x * 8 + wave;         // wave-uniform
    if (grp >= numGrp) return;                         // whole-wave skip: EXEC stays all-1s

    const int tm  = grp / groupsN;
    const int gn  = grp % groupsN;
    const int l15 = lane & 15;
    const int hi  = lane >> 4;
    const int rowA = tm * 16 + l15;                    // A-fragment row for this lane

    v8f acc0 = {}, acc1 = {}, acc2 = {}, acc3 = {};
#pragma unroll
    for (int kk = 0; kk < 4; ++kk) {
        const int kb = kk * 32;
        // A 16x32 f16 layout: element e -> k = kb + 16*(e>>3) + 8*hi + (e&7)
        const float* ap = A + (size_t)rowA * CAT + kb + 8 * hi;
        v16h a;
#pragma unroll
        for (int e = 0; e < 8; ++e) {
            a[e]     = (_Float16)ap[e];
            a[e + 8] = (_Float16)ap[e + 16];
        }
#pragma unroll
        for (int j = 0; j < 4; ++j) {
            // B 32x16 f16 layout: element e -> k = kb + 16*(e>>3) + 8*hi + (e&7)
            const int colB = (gn * 4 + j) * 16 + l15;
            const float* bp = B + (size_t)colB * CAT + kb + 8 * hi;
            v16h b;
#pragma unroll
            for (int e = 0; e < 16; ++e) b[e] = (_Float16)bp[e + 8 * (e >> 3)];
            v8f& acc = (j == 0) ? acc0 : (j == 1) ? acc1 : (j == 2) ? acc2 : acc3;
            acc = __builtin_amdgcn_wmma_f32_16x16x32_f16(
                /*neg_a=*/false, a, /*neg_b=*/false, b,
                /*c_mod=*/(short)0, acc, /*reuse_a=*/false, /*reuse_b=*/false);
        }
    }

#pragma unroll
    for (int j = 0; j < 4; ++j) {
        const int col = (gn * 4 + j) * 16 + l15;
        const float bv = bias ? bias[col] : 0.f;       // single lane-uniform-per-strip load
        const v8f acc = (j == 0) ? acc0 : (j == 1) ? acc1 : (j == 2) ? acc2 : acc3;
#pragma unroll
        for (int r = 0; r < 8; ++r) {
            const int row = tm * 16 + r + 8 * hi;      // C/D layout: VGPR r -> row r+8*hi
            C[(size_t)row * Nc + col] = acc[r] + bv;
        }
    }
}

// ---------------------------------------------------------------------------
// Attention scores: s_src[n,h] = sum_d hW[n,h*32+d]*a_src[h,d]  (same for dst)
// ---------------------------------------------------------------------------
__global__ __launch_bounds__(256) void attn_score_kernel(
    const float* __restrict__ hW, const float* __restrict__ a_src,
    const float* __restrict__ a_dst, float* __restrict__ s_src,
    float* __restrict__ s_dst, int N)
{
    const int i = blockIdx.x * 256 + threadIdx.x;
    if (i >= N * HEADS) return;
    const int n = i >> 2, h = i & 3;
    const float* hp = hW + (size_t)n * CAT + h * HID;
    const float* as = a_src + h * HID;
    const float* ad = a_dst + h * HID;
    float ss = 0.f, sd = 0.f;
#pragma unroll
    for (int d = 0; d < HID; ++d) { const float v = hp[d]; ss += v * as[d]; sd += v * ad[d]; }
    s_src[i] = ss; s_dst[i] = sd;
}

// ---------------------------------------------------------------------------
// Edge helpers / passes (Etot = E real edges + N self loops)
// ---------------------------------------------------------------------------
__device__ __forceinline__ void edge_nodes(const int* __restrict__ ei, int E, int t,
                                           int& s, int& d) {
    if (t < E) { s = ei[t]; d = ei[E + t]; } else { s = t - E; d = s; }
}

__global__ __launch_bounds__(256) void edge_max_kernel(
    const int* __restrict__ ei, int E, int Etot,
    const float* __restrict__ s_src, const float* __restrict__ s_dst,
    unsigned* __restrict__ m_u)
{
    const int t = blockIdx.x * 256 + threadIdx.x;
    if (t >= Etot) return;
    int s, d; edge_nodes(ei, E, t, s, d);
    const float4 a = *reinterpret_cast<const float4*>(s_src + 4 * (size_t)s);
    const float4 b = *reinterpret_cast<const float4*>(s_dst + 4 * (size_t)d);
    unsigned* mp = m_u + 4 * (size_t)d;
    atomicMax(mp + 0, enc_ord(lrelu(a.x + b.x)));
    atomicMax(mp + 1, enc_ord(lrelu(a.y + b.y)));
    atomicMax(mp + 2, enc_ord(lrelu(a.z + b.z)));
    atomicMax(mp + 3, enc_ord(lrelu(a.w + b.w)));
}

__global__ __launch_bounds__(256) void edge_denom_kernel(
    const int* __restrict__ ei, int E, int Etot,
    const float* __restrict__ s_src, const float* __restrict__ s_dst,
    const unsigned* __restrict__ m_u, float* __restrict__ denom)
{
    const int t = blockIdx.x * 256 + threadIdx.x;
    if (t >= Etot) return;
    int s, d; edge_nodes(ei, E, t, s, d);
    const float4 a = *reinterpret_cast<const float4*>(s_src + 4 * (size_t)s);
    const float4 b = *reinterpret_cast<const float4*>(s_dst + 4 * (size_t)d);
    const unsigned* mp = m_u + 4 * (size_t)d;
    float* dp = denom + 4 * (size_t)d;
    atomicAdd(dp + 0, expf(lrelu(a.x + b.x) - dec_ord(mp[0])));
    atomicAdd(dp + 1, expf(lrelu(a.y + b.y) - dec_ord(mp[1])));
    atomicAdd(dp + 2, expf(lrelu(a.z + b.z) - dec_ord(mp[2])));
    atomicAdd(dp + 3, expf(lrelu(a.w + b.w) - dec_ord(mp[3])));
}

// 32 lanes per edge: lane = channel-in-head; each lane handles all 4 heads.
__global__ __launch_bounds__(256) void edge_aggr_kernel(
    const int* __restrict__ ei, int E, int Etot,
    const float* __restrict__ s_src, const float* __restrict__ s_dst,
    const unsigned* __restrict__ m_u, const float* __restrict__ denom,
    const float* __restrict__ hW, float* __restrict__ agg)
{
    const long gid = (long)blockIdx.x * 256 + threadIdx.x;
    const int t   = (int)(gid >> 5);
    const int dch = (int)(gid & 31);
    if (t >= Etot) return;
    int s, d; edge_nodes(ei, E, t, s, d);
    const float4 a = *reinterpret_cast<const float4*>(s_src + 4 * (size_t)s);
    const float4 b = *reinterpret_cast<const float4*>(s_dst + 4 * (size_t)d);
    const unsigned* mp = m_u + 4 * (size_t)d;
    const float* dp = denom + 4 * (size_t)d;
    const float al0 = expf(lrelu(a.x + b.x) - dec_ord(mp[0])) / dp[0];
    const float al1 = expf(lrelu(a.y + b.y) - dec_ord(mp[1])) / dp[1];
    const float al2 = expf(lrelu(a.z + b.z) - dec_ord(mp[2])) / dp[2];
    const float al3 = expf(lrelu(a.w + b.w) - dec_ord(mp[3])) / dp[3];
    const float* hp = hW + (size_t)s * CAT + dch;
    float* op = agg + (size_t)d * CAT + dch;
    atomicAdd(op + 0,        al0 * hp[0]);
    atomicAdd(op + 1 * HID,  al1 * hp[1 * HID]);
    atomicAdd(op + 2 * HID,  al2 * hp[2 * HID]);
    atomicAdd(op + 3 * HID,  al3 * hp[3 * HID]);
}

// ---------------------------------------------------------------------------
// bias + ELU (in place)
// ---------------------------------------------------------------------------
__global__ __launch_bounds__(256) void bias_elu_kernel(
    float* __restrict__ x, const float* __restrict__ bias, long total)
{
    const long i = (long)blockIdx.x * 256 + threadIdx.x;
    if (i >= total) return;
    const float v = x[i] + bias[i & (CAT - 1)];
    x[i] = v > 0.f ? v : (expf(v) - 1.f);
}

// ---------------------------------------------------------------------------
// GraphNorm: per-graph mean/var via atomics, then normalize
// ---------------------------------------------------------------------------
__global__ __launch_bounds__(256) void gn_sum_kernel(
    const float* __restrict__ x, const int* __restrict__ batch,
    float* __restrict__ gsum, float* __restrict__ gcnt, int N)
{
    const long i = (long)blockIdx.x * 256 + threadIdx.x;
    if (i >= (long)N * CAT) return;
    const int n = (int)(i >> 7), c = (int)(i & (CAT - 1));
    const int g = batch[n];
    atomicAdd(&gsum[(size_t)g * CAT + c], x[i]);
    if (c == 0) atomicAdd(&gcnt[g], 1.0f);
}

__global__ __launch_bounds__(256) void gn_var_kernel(
    const float* __restrict__ x, const int* __restrict__ batch,
    const float* __restrict__ gsum, const float* __restrict__ gcnt,
    const float* __restrict__ ms, float* __restrict__ gsumsq, int N)
{
    const long i = (long)blockIdx.x * 256 + threadIdx.x;
    if (i >= (long)N * CAT) return;
    const int n = (int)(i >> 7), c = (int)(i & (CAT - 1));
    const int g = batch[n];
    const float mean = gsum[(size_t)g * CAT + c] / gcnt[g];
    const float sub  = x[i] - ms[c] * mean;
    atomicAdd(&gsumsq[(size_t)g * CAT + c], sub * sub);
}

__global__ __launch_bounds__(256) void gn_apply_kernel(
    const float* __restrict__ x, const int* __restrict__ batch,
    const float* __restrict__ gsum, const float* __restrict__ gsumsq,
    const float* __restrict__ gcnt, const float* __restrict__ w,
    const float* __restrict__ b, const float* __restrict__ ms,
    float* __restrict__ y, int N)
{
    const long i = (long)blockIdx.x * 256 + threadIdx.x;
    if (i >= (long)N * CAT) return;
    const int n = (int)(i >> 7), c = (int)(i & (CAT - 1));
    const int g = batch[n];
    const float cnt  = gcnt[g];
    const float mean = gsum[(size_t)g * CAT + c] / cnt;
    const float var  = gsumsq[(size_t)g * CAT + c] / cnt;
    const float sub  = x[i] - ms[c] * mean;
    y[i] = w[c] * sub * rsqrtf(var + GN_EPS) + b[c];
}

// ---------------------------------------------------------------------------
// Global max pool per graph + decode
// ---------------------------------------------------------------------------
__global__ __launch_bounds__(256) void pool_max_kernel(
    const float* __restrict__ x, const int* __restrict__ batch,
    unsigned* __restrict__ pu, int N)
{
    const long i = (long)blockIdx.x * 256 + threadIdx.x;
    if (i >= (long)N * CAT) return;
    const int n = (int)(i >> 7), c = (int)(i & (CAT - 1));
    atomicMax(&pu[(size_t)batch[n] * CAT + c], enc_ord(x[i]));
}

__global__ __launch_bounds__(256) void decode_kernel(
    const unsigned* __restrict__ pu, float* __restrict__ pf, int n)
{
    const int i = blockIdx.x * 256 + threadIdx.x;
    if (i < n) pf[i] = dec_ord(pu[i]);
}

__global__ __launch_bounds__(256) void fill_u32_kernel(
    unsigned* __restrict__ p, unsigned v, long n)
{
    const long i = (long)blockIdx.x * 256 + threadIdx.x;
    if (i < n) p[i] = v;
}

// ---------------------------------------------------------------------------
// Host orchestration
// ---------------------------------------------------------------------------
static inline int cdiv(long a, long b) { return (int)((a + b - 1) / b); }

extern "C" void kernel_launch(void* const* d_in, const int* in_sizes, int n_in,
                              void* d_out, int out_size, void* d_ws, size_t ws_size,
                              hipStream_t stream)
{
    const float* x     = (const float*)d_in[0];
    const int*   ei    = (const int*)d_in[1];
    const int*   batch = (const int*)d_in[2];
    const float* enc_w = (const float*)d_in[3];
    const float* enc_b = (const float*)d_in[4];
    const float* w1    = (const float*)d_in[5];
    const float* as1   = (const float*)d_in[6];
    const float* ad1   = (const float*)d_in[7];
    const float* b1    = (const float*)d_in[8];
    const float* n1w   = (const float*)d_in[9];
    const float* n1b   = (const float*)d_in[10];
    const float* n1ms  = (const float*)d_in[11];
    const float* w2    = (const float*)d_in[12];
    const float* as2   = (const float*)d_in[13];
    const float* ad2   = (const float*)d_in[14];
    const float* b2    = (const float*)d_in[15];
    const float* n2w   = (const float*)d_in[16];
    const float* n2b   = (const float*)d_in[17];
    const float* n2ms  = (const float*)d_in[18];
    const float* fcw   = (const float*)d_in[19];
    const float* fcb   = (const float*)d_in[20];
    float* out = (float*)d_out;
    (void)n_in; (void)out_size; (void)ws_size;

    const int N    = in_sizes[0] / CAT;   // 100000
    const int E    = in_sizes[1] / 2;     // 1600000
    const int Etot = E + N;               // + self loops
    const int G    = NGRAPH;

    // ---- workspace layout (floats) ----
    float* ws = (float*)d_ws;
    size_t o = 0;
    float*    buf0 = ws + o; o += (size_t)N * CAT;   // features
    float*    buf1 = ws + o; o += (size_t)N * CAT;   // hW
    float*    buf2 = ws + o; o += (size_t)N * CAT;   // aggregation
    float*    ssrc = ws + o; o += (size_t)N * HEADS;
    float*    sdst = ws + o; o += (size_t)N * HEADS;
    unsigned* mu   = (unsigned*)(ws + o); o += (size_t)N * HEADS;
    float*    den  = ws + o; o += (size_t)N * HEADS;
    float*    gsum = ws + o; o += (size_t)G * CAT;   // gsum/gsq/gcnt contiguous
    float*    gsq  = ws + o; o += (size_t)G * CAT;
    float*    gcnt = ws + o; o += (size_t)G;
    unsigned* pu   = (unsigned*)(ws + o); o += (size_t)G * CAT;
    float*    pf   = ws + o; o += (size_t)G * CAT;

    auto gemm = [&](const float* A, const float* B, const float* bias, float* C,
                    int M, int Nc) {
        const int groups = (M / 16) * (Nc / 64);       // 16x64 strip per wave
        wmma_gemm_nt_kernel<<<cdiv(groups, 8), 256, 0, stream>>>(A, B, bias, C, M, Nc);
    };
    auto fill = [&](void* p, unsigned v, long n) {
        fill_u32_kernel<<<cdiv(n, 256), 256, 0, stream>>>((unsigned*)p, v, n);
    };

    auto gat_layer = [&](const float* feat, const float* W, const float* asv,
                         const float* adv, const float* bias, const float* gw,
                         const float* gb, const float* gms, float* feat_out) {
        gemm(feat, W, nullptr, buf1, N, CAT);                            // hW = feat @ W.T
        attn_score_kernel<<<cdiv((long)N * HEADS, 256), 256, 0, stream>>>(
            buf1, asv, adv, ssrc, sdst, N);
        fill(mu,   0u, (long)N * HEADS);          // 0 == enc_ord(-inf floor)
        fill(den,  0u, (long)N * HEADS);
        fill(buf2, 0u, (long)N * CAT);
        edge_max_kernel<<<cdiv(Etot, 256), 256, 0, stream>>>(ei, E, Etot, ssrc, sdst, mu);
        edge_denom_kernel<<<cdiv(Etot, 256), 256, 0, stream>>>(ei, E, Etot, ssrc, sdst, mu, den);
        edge_aggr_kernel<<<cdiv((long)Etot * 32, 256), 256, 0, stream>>>(
            ei, E, Etot, ssrc, sdst, mu, den, buf1, buf2);
        bias_elu_kernel<<<cdiv((long)N * CAT, 256), 256, 0, stream>>>(buf2, bias, (long)N * CAT);
        fill(gsum, 0u, (long)G * CAT * 2 + G);    // gsum, gsq, gcnt contiguous
        gn_sum_kernel<<<cdiv((long)N * CAT, 256), 256, 0, stream>>>(buf2, batch, gsum, gcnt, N);
        gn_var_kernel<<<cdiv((long)N * CAT, 256), 256, 0, stream>>>(
            buf2, batch, gsum, gcnt, gms, gsq, N);
        gn_apply_kernel<<<cdiv((long)N * CAT, 256), 256, 0, stream>>>(
            buf2, batch, gsum, gsq, gcnt, gw, gb, gms, feat_out, N);
    };

    // ---- pipeline ----
    gemm(x, enc_w, enc_b, buf0, N, CAT);                                 // encoder
    gat_layer(buf0, w1, as1, ad1, b1, n1w, n1b, n1ms, buf0);             // GAT1 + norm
    gat_layer(buf0, w2, as2, ad2, b2, n2w, n2b, n2ms, buf0);             // GAT2 + norm

    fill(pu, 0u, (long)G * CAT);
    pool_max_kernel<<<cdiv((long)N * CAT, 256), 256, 0, stream>>>(buf0, batch, pu, N);
    decode_kernel<<<cdiv((long)G * CAT, 256), 256, 0, stream>>>(pu, pf, G * CAT);

    gemm(pf, fcw, fcb, out, G, 64);                                      // final FC -> d_out
}
